// RT_79706003079219
// MI455X (gfx1250) — matrix operand order repaired and checked by reading, not compiled, not run on hardware
//
#include <hip/hip_runtime.h>

// ---------------- types & helpers ----------------
typedef __attribute__((ext_vector_type(16))) __bf16 v16bf;
typedef __attribute__((ext_vector_type(8)))  __bf16 v8bf;
typedef __attribute__((ext_vector_type(8)))  float  v8f;

__device__ __forceinline__ unsigned short f2bf(float f) {
  union { float f; unsigned u; } a; a.f = f;
  unsigned r = a.u + 0x7FFFu + ((a.u >> 16) & 1u);   // round-to-nearest-even
  return (unsigned short)(r >> 16);
}

__device__ __forceinline__ v16bf joinbf(v8bf lo, v8bf hi) {
  return __builtin_shufflevector(lo, hi, 0,1,2,3,4,5,6,7,8,9,10,11,12,13,14,15);
}

__device__ __forceinline__ v8f wmma_bf16(v16bf a, v16bf b, v8f c) {
  return __builtin_amdgcn_wmma_f32_16x16x32_bf16(false, a, false, b, (short)0, c, false, false);
}

// ---------------- small utility kernels ----------------
__global__ void cvt_f32_bf16(const float* __restrict__ x, unsigned short* __restrict__ y, int n) {
  int i = blockIdx.x * 256 + threadIdx.x;
  if (i < n) y[i] = f2bf(x[i]);
}

// Pack q & v rows out of a [192,64] qkv weight (per head h: q rows 48h..48h+15,
// v rows 48h+32..48h+47) into [128,64] bf16 + packed f32 bias.
__global__ void pack_qv(const float* __restrict__ Wsrc, const float* __restrict__ bsrc,
                        unsigned short* __restrict__ Wout, float* __restrict__ bout) {
  int idx = blockIdx.x * 256 + threadIdx.x;          // 128*64 = 8192
  if (idx < 128 * 64) {
    int po = idx >> 6, k = idx & 63;
    int h = po >> 5, r = po & 31;
    int src = (r < 16) ? (48 * h + r) : (48 * h + 32 + (r - 16));
    Wout[idx] = f2bf(Wsrc[src * 64 + k]);
    if (k == 0) bout[po] = bsrc[src];
  }
}

// y = LN(x + d) over last dim 64; writes f32 and optional bf16 copy.
__global__ void residual_ln64(const float* __restrict__ X, const float* __restrict__ Dlt,
                              const float* __restrict__ g, const float* __restrict__ bt,
                              float* __restrict__ Yf, unsigned short* __restrict__ Yh) {
  __shared__ float s[64];
  int r = blockIdx.x, d = threadIdx.x;
  size_t idx = (size_t)r * 64 + d;
  float v = X[idx] + Dlt[idx];
  s[d] = v; __syncthreads();
  for (int off = 32; off > 0; off >>= 1) { if (d < off) s[d] += s[d + off]; __syncthreads(); }
  float mean = s[0] * (1.0f / 64.0f);
  __syncthreads();
  float c = v - mean;
  s[d] = c * c; __syncthreads();
  for (int off = 32; off > 0; off >>= 1) { if (d < off) s[d] += s[d + off]; __syncthreads(); }
  float var = s[0] * (1.0f / 64.0f);
  float y = c * rsqrtf(var + 1e-5f) * g[d] + bt[d];
  Yf[idx] = y;
  if (Yh) Yh[idx] = f2bf(y);
}

// ---------------- bf16 WMMA GEMM: Y[R,O] = act(A[R,K] @ W[O,K]^T + b) ----------------
// Register-blocked: each wave owns a 16x64 output strip (4 O-tiles), loading the A
// fragment once per K-step and reusing it across 4 WMMAs. K is a template constant so
// the K-loop fully unrolls: every fragment gets distinct vregs and the scheduler can
// hoist load clauses over earlier WMMAs (partial s_wait_loadcnt instead of full drains).
// blockDim=128: wave w -> row-tile rt = blockIdx.x*4+w; grid=(R/64, O/64).
// A fragment per ISA layout: lane m=lane&15, K-halves by lane>>4, 8-half (16B) loads.
// B fragment: lane n=lane&15 holds W row (o_base+n), same K-halves layout as A.
template<int K, bool RELU>
__global__ void gemm_bf16_k(const unsigned short* __restrict__ A,
                            const unsigned short* __restrict__ Wb,
                            const float* __restrict__ bias,
                            float* __restrict__ Yf,
                            unsigned short* __restrict__ Yh,
                            int O) {
  const int lane = threadIdx.x & 31;
  const int wave = threadIdx.x >> 5;
  const int nn = lane & 15;
  const int kh = lane >> 4;
  const int rt  = blockIdx.x * 4 + wave;
  const int ot0 = blockIdx.y * 4;
  const size_t arow  = (size_t)(rt * 16 + nn) * K;
  const size_t wrow0 = (size_t)(ot0 * 16 + nn) * K;
  v8f c[4] = {};
#pragma unroll
  for (int kb = 0; kb < K; kb += 32) {
    v8bf a0 = *(const v8bf*)(A + arow + kb + kh * 8);
    v8bf a1 = *(const v8bf*)(A + arow + kb + 16 + kh * 8);
    v16bf av = joinbf(a0, a1);
    v16bf bv[4];
#pragma unroll
    for (int t = 0; t < 4; ++t) {
      const size_t wr = wrow0 + (size_t)t * 16 * K;
      bv[t] = joinbf(*(const v8bf*)(Wb + wr + kb + kh * 8),
                     *(const v8bf*)(Wb + wr + kb + 16 + kh * 8));
    }
#pragma unroll
    for (int t = 0; t < 4; ++t) c[t] = wmma_bf16(av, bv[t], c[t]);
  }
#pragma unroll
  for (int t = 0; t < 4; ++t) {
    const int oc = (ot0 + t) * 16 + nn;
    const float bvv = bias[oc];
#pragma unroll
    for (int v = 0; v < 8; ++v) {
      int m = v + 8 * kh;                // D: lanes 0-15 -> M=v, lanes 16-31 -> M=v+8
      float val = c[t][v] + bvv;
      if (RELU) val = fmaxf(val, 0.0f);
      size_t yi = (size_t)(rt * 16 + m) * O + oc;
      if (Yf) Yf[yi] = val;
      if (Yh) Yh[yi] = f2bf(val);
    }
  }
}

// Gather-GEMM over edge-space rows r=(b,i,j) (B=4,N=256): the K dim is a concat of
// 64-wide segments pulled straight from edge/node bf16 buffers (no cat materialization).
// MODE 0: K=128 [node_j | node_i]   (init_edge input)
// MODE 1: K=256 [edge | edge_rev | node_j | node_i]  (e1a input)
// Same 16x64-per-wave register blocking + staged B fragments; O=128 -> grid=(R/64, 2).
template<int MODE, bool RELU>
__global__ void gemm_gather_k(const unsigned short* __restrict__ Ebf,
                              const unsigned short* __restrict__ Nbf,
                              const unsigned short* __restrict__ Wb,
                              const float* __restrict__ bias,
                              float* __restrict__ Yf,
                              unsigned short* __restrict__ Yh,
                              int O) {
  const int K = (MODE == 0) ? 128 : 256;
  const int lane = threadIdx.x & 31;
  const int wave = threadIdx.x >> 5;
  const int nn = lane & 15;
  const int kh = lane >> 4;
  const int rt  = blockIdx.x * 4 + wave;
  const int ot0 = blockIdx.y * 4;
  const int row = rt * 16 + nn;
  const int b = row >> 16, i = (row >> 8) & 255, j = row & 255;
  const unsigned short* seg[4];
  if (MODE == 0) {
    seg[0] = Nbf + (size_t)((b << 8) | j) * 64;   // recv = node[j]
    seg[1] = Nbf + (size_t)((b << 8) | i) * 64;   // send = node[i]
    seg[2] = seg[0]; seg[3] = seg[1];
  } else {
    seg[0] = Ebf + (size_t)row * 64;                            // edge[b,i,j]
    seg[1] = Ebf + (size_t)((b << 16) | (j << 8) | i) * 64;     // rev = edge[b,j,i]
    seg[2] = Nbf + (size_t)((b << 8) | j) * 64;                 // src = node[j]
    seg[3] = Nbf + (size_t)((b << 8) | i) * 64;                 // tgt = node[i]
  }
  const size_t wrow0 = (size_t)(ot0 * 16 + nn) * K;
  v8f c[4] = {};
#pragma unroll
  for (int kb = 0; kb < K; kb += 32) {          // unrolled -> seg index is constant
    int k0 = kb + kh * 8;
    int k1 = kb + 16 + kh * 8;
    v8bf a0 = *(const v8bf*)(seg[kb >> 6] + (k0 & 63));
    v8bf a1 = *(const v8bf*)(seg[kb >> 6] + (k1 & 63));
    v16bf av = joinbf(a0, a1);
    v16bf bv[4];
#pragma unroll
    for (int t = 0; t < 4; ++t) {
      const size_t wr = wrow0 + (size_t)t * 16 * K;
      bv[t] = joinbf(*(const v8bf*)(Wb + wr + kb + kh * 8),
                     *(const v8bf*)(Wb + wr + kb + 16 + kh * 8));
    }
#pragma unroll
    for (int t = 0; t < 4; ++t) c[t] = wmma_bf16(av, bv[t], c[t]);
  }
#pragma unroll
  for (int t = 0; t < 4; ++t) {
    const int oc = (ot0 + t) * 16 + nn;
    const float bvv = bias[oc];
#pragma unroll
    for (int v = 0; v < 8; ++v) {
      int m = v + 8 * kh;
      float val = c[t][v] + bvv;
      if (RELU) val = fmaxf(val, 0.0f);
      size_t yi = (size_t)(rt * 16 + m) * O + oc;
      if (Yf) Yf[yi] = val;
      if (Yh) Yh[yi] = f2bf(val);
    }
  }
}

// ---------------- attention (faithful: scores = ||q_n[i]+q_e[i,j]||^2 * scale) ----------------
// qvn: [B*N,128] packed per head h: q at 32h..32h+15, v at 32h+16..32h+31
// qve: [B*N*N,128] same packing. out: bf16 [B*N,64], channel = h*16+d.
__global__ void attention_kernel(const float* __restrict__ qvn, const float* __restrict__ qve,
                                 unsigned short* __restrict__ out) {
  const int i = blockIdx.x, h = blockIdx.y, b = blockIdx.z;
  const int j = threadIdx.x;                 // 256
  __shared__ float qn[16];
  __shared__ float p[256];
  __shared__ float red[256];
  __shared__ float part[16][17];
  const size_t rowN = (size_t)(b * 256 + i);
  if (j < 16) qn[j] = qvn[rowN * 128 + h * 32 + j];
  __syncthreads();
  const float* qe = qve + (rowN * 256 + j) * 128 + h * 32;
  float s = 0.0f;
#pragma unroll
  for (int d = 0; d < 16; ++d) { float t = qn[d] + qe[d]; s += t * t; }
  s *= 0.25f;                                // 1/sqrt(16)
  red[j] = s; __syncthreads();
  for (int off = 128; off > 0; off >>= 1) { if (j < off) red[j] = fmaxf(red[j], red[j + off]); __syncthreads(); }
  float mx = red[0]; __syncthreads();
  float e = __expf(s - mx);
  p[j] = e; red[j] = e; __syncthreads();
  for (int off = 128; off > 0; off >>= 1) { if (j < off) red[j] += red[j + off]; __syncthreads(); }
  float inv = 1.0f / red[0];
  __syncthreads();
  const int d = j & 15, grp = j >> 4;        // 16 groups x 16 dims
  float acc = 0.0f;
  for (int jj = grp * 16; jj < grp * 16 + 16; ++jj) {
    float vn = qvn[(size_t)(b * 256 + jj) * 128 + h * 32 + 16 + d];
    float ve = qve[(rowN * 256 + jj) * 128 + h * 32 + 16 + d];
    acc += p[jj] * (vn + ve);
  }
  part[d][grp] = acc;
  __syncthreads();
  if (j < 16) {
    float s2 = 0.0f;
#pragma unroll
    for (int gg = 0; gg < 16; ++gg) s2 += part[j][gg];
    out[rowN * 64 + h * 16 + j] = f2bf(s2 * inv);
  }
}

// ---------------- host ----------------
extern "C" void kernel_launch(void* const* d_in, const int* in_sizes, int n_in,
                              void* d_out, int out_size, void* d_ws, size_t ws_size,
                              hipStream_t stream) {
  (void)in_sizes; (void)n_in; (void)out_size; (void)ws_size;
  const int R  = 4 * 256 * 256;   // edge-space rows = 262144
  const int Rn = 4 * 256;         // node-space rows = 1024
  const float* node_in = (const float*)d_in[0];

  // ---- workspace layout ----
  char* base = (char*)d_ws;
  float*          edge   = (float*)(base);                                    // 67,108,864 B
  unsigned short* edgebf = (unsigned short*)(base + (size_t)67108864);        // 33,554,432 B
  char*           big    = base + (size_t)100663296;                          // 134,217,728 B (time-shared)
  float*          qve    = (float*)big;                                       // attention phase
  unsigned short* hbf    = (unsigned short*)big;                              // edge-MLP phase
  float*          etmp   = (float*)(big + (size_t)67108864);
  char* sm = base + (size_t)234881024;
  float*          nodeA   = (float*)sm;          sm += 262144;
  unsigned short* nodebf  = (unsigned short*)sm; sm += 131072;
  float*          qvn     = (float*)sm;          sm += 524288;
  unsigned short* attnout = (unsigned short*)sm; sm += 131072;
  float*          attnproj= (float*)sm;          sm += 262144;
  unsigned short* hn      = (unsigned short*)sm; sm += 262144;
  float*          ffout   = (float*)sm;          sm += 262144;
  auto alloc = [&](size_t bytes) -> void* { void* p = sm; sm += (bytes + 511) & ~(size_t)511; return p; };

  auto cvtw = [&](int idx, int n) -> unsigned short* {
    unsigned short* p = (unsigned short*)alloc((size_t)n * 2);
    cvt_f32_bf16<<<dim3((n + 255) / 256), dim3(256), 0, stream>>>((const float*)d_in[idx], p, n);
    return p;
  };

  // ---- convert / pack weights (every call: deterministic, capture-safe) ----
  unsigned short* n2e1w = cvtw(2, 128 * 128);
  unsigned short* n2e2w = cvtw(4, 64 * 128);
  unsigned short *qvnW[3], *qveW[3], *oW[3], *f1W[3], *f2W[3], *e1aW[3], *e1bW[3], *e2aW[3], *e2bW[3];
  float *qvnB[3], *qveB[3];
  for (int l = 0; l < 3; ++l) {
    int bs = 6 + 26 * l;
    qvnW[l] = (unsigned short*)alloc(128 * 64 * 2); qvnB[l] = (float*)alloc(128 * 4);
    pack_qv<<<dim3(32), dim3(256), 0, stream>>>((const float*)d_in[bs + 0], (const float*)d_in[bs + 1], qvnW[l], qvnB[l]);
    qveW[l] = (unsigned short*)alloc(128 * 64 * 2); qveB[l] = (float*)alloc(128 * 4);
    pack_qv<<<dim3(32), dim3(256), 0, stream>>>((const float*)d_in[bs + 2], (const float*)d_in[bs + 3], qveW[l], qveB[l]);
    oW[l]   = cvtw(bs + 4,  64 * 64);
    f1W[l]  = cvtw(bs + 6,  128 * 64);
    f2W[l]  = cvtw(bs + 8,  64 * 128);
    e1aW[l] = cvtw(bs + 14, 128 * 256);
    e1bW[l] = cvtw(bs + 16, 64 * 128);
    e2aW[l] = cvtw(bs + 18, 128 * 64);
    e2bW[l] = cvtw(bs + 20, 64 * 128);
  }

  auto gemm = [&](const unsigned short* A, const unsigned short* Wb, const float* bias,
                  float* Yf, unsigned short* Yh, int Rr, int K, int O, bool relu) {
    dim3 g(Rr / 64, O / 64), blk(128);
    if (K == 64) {
      if (relu) gemm_bf16_k<64, true ><<<g, blk, 0, stream>>>(A, Wb, bias, Yf, Yh, O);
      else      gemm_bf16_k<64, false><<<g, blk, 0, stream>>>(A, Wb, bias, Yf, Yh, O);
    } else {
      if (relu) gemm_bf16_k<128, true ><<<g, blk, 0, stream>>>(A, Wb, bias, Yf, Yh, O);
      else      gemm_bf16_k<128, false><<<g, blk, 0, stream>>>(A, Wb, bias, Yf, Yh, O);
    }
  };

  // ---- node init ----
  hipMemcpyAsync(nodeA, node_in, (size_t)Rn * 64 * 4, hipMemcpyDeviceToDevice, stream);
  cvt_f32_bf16<<<dim3((Rn * 64 + 255) / 256), dim3(256), 0, stream>>>(node_in, nodebf, Rn * 64);

  // ---- init_edge: relu(lin(cat(node_j,node_i))) -> lin -> edge (f32 + bf16) ----
  gemm_gather_k<0, true><<<dim3(R / 64, 2), dim3(128), 0, stream>>>(
      edgebf, nodebf, n2e1w, (const float*)d_in[3], nullptr, hbf, 128);
  gemm(hbf, n2e2w, (const float*)d_in[5], edge, edgebf, R, 128, 64, false);

  // ---- layers ----
  for (int l = 0; l < 3; ++l) {
    int bs = 6 + 26 * l;
    // attention
    gemm(nodebf, qvnW[l], qvnB[l], qvn, nullptr, Rn, 64, 128, false);
    gemm(edgebf, qveW[l], qveB[l], qve, nullptr, R, 64, 128, false);
    attention_kernel<<<dim3(256, 4, 4), dim3(256), 0, stream>>>(qvn, qve, attnout);
    gemm(attnout, oW[l], (const float*)d_in[bs + 5], attnproj, nullptr, Rn, 64, 64, false);
    residual_ln64<<<dim3(Rn), dim3(64), 0, stream>>>(nodeA, attnproj,
        (const float*)d_in[bs + 10], (const float*)d_in[bs + 11], nodeA, nodebf);
    // node FFN
    gemm(nodebf, f1W[l], (const float*)d_in[bs + 7], nullptr, hn, Rn, 64, 128, true);
    gemm(hn, f2W[l], (const float*)d_in[bs + 9], ffout, nullptr, Rn, 128, 64, false);
    residual_ln64<<<dim3(Rn), dim3(64), 0, stream>>>(nodeA, ffout,
        (const float*)d_in[bs + 12], (const float*)d_in[bs + 13], nodeA, nodebf);
    // edge update 1: cat(edge,rev,src,tgt) -> relu lin -> lin -> +res -> LN
    gemm_gather_k<1, true><<<dim3(R / 64, 2), dim3(128), 0, stream>>>(
        edgebf, nodebf, e1aW[l], (const float*)d_in[bs + 15], nullptr, hbf, 128);
    gemm(hbf, e1bW[l], (const float*)d_in[bs + 17], etmp, nullptr, R, 128, 64, false);
    residual_ln64<<<dim3(R), dim3(64), 0, stream>>>(edge, etmp,
        (const float*)d_in[bs + 22], (const float*)d_in[bs + 23], edge, edgebf);
    // edge update 2
    gemm(edgebf, e2aW[l], (const float*)d_in[bs + 19], nullptr, hbf, R, 64, 128, true);
    gemm(hbf, e2bW[l], (const float*)d_in[bs + 21], etmp, nullptr, R, 128, 64, false);
    residual_ln64<<<dim3(R), dim3(64), 0, stream>>>(edge, etmp,
        (const float*)d_in[bs + 24], (const float*)d_in[bs + 25], edge, edgebf);
  }

  hipMemcpyAsync(d_out, nodeA, (size_t)Rn * 64 * 4, hipMemcpyDeviceToDevice, stream);
}
